// Predict_Attend_85555748536829
// MI455X (gfx1250) — matrix operand order repaired and checked by reading, not compiled, not run on hardware
//
#include <hip/hip_runtime.h>

// Problem constants from the reference: (b, h, n, d) = (2, 16, 2048, 64), all fp32.
#define BATCH 2
#define HEADS 16
#define SEQ   2048
#define DIM   64

typedef __attribute__((ext_vector_type(16))) __bf16 v16bf;
typedef __attribute__((ext_vector_type(8)))  float  v8f;

union Frag16 {
  unsigned short u[16];
  v16bf          v;
  uint4          q[2];
};

// f32 -> bf16 using the native hardware conversion.
__device__ __forceinline__ unsigned short f2bf(float f) {
  __bf16 b = (__bf16)f;
  return __builtin_bit_cast(unsigned short, b);
}

// Packed f32x2 -> bf16x2 (lo in [15:0], hi in [31:16]).
__device__ __forceinline__ unsigned pk_bf16(float lo, float hi) {
  unsigned d;
  asm("v_cvt_pk_bf16_f32 %0, %1, %2" : "=v"(d) : "v"(lo), "v"(hi));
  return d;
}

// Packed bf16x2 max (VOP3P V_PK_MAX_NUM_BF16).
__device__ __forceinline__ unsigned pk_max_bf16(unsigned a, unsigned b) {
  unsigned d;
  asm("v_pk_max_num_bf16 %0, %1, %2" : "=v"(d) : "v"(a), "v"(b));
  return d;
}

// Load 8 contiguous floats, scale, pack to 8 bf16.
__device__ __forceinline__ void pack8s(const float* p, unsigned short* d, float s) {
  float4 a = *(const float4*)(p);
  float4 b = *(const float4*)(p + 4);
  d[0] = f2bf(a.x * s); d[1] = f2bf(a.y * s); d[2] = f2bf(a.z * s); d[3] = f2bf(a.w * s);
  d[4] = f2bf(b.x * s); d[5] = f2bf(b.y * s); d[6] = f2bf(b.z * s); d[7] = f2bf(b.w * s);
}

// ---------------- prep kernel 1: f32 -> bf16, same layout (keys_o) ----------------
__global__ __launch_bounds__(256)
void cvt_bf16_kernel(const float* __restrict__ src, unsigned short* __restrict__ dst) {
  const int idx = blockIdx.x * 256 + threadIdx.x;     // one uint4 (8 bf16) per thread
  const float4* s = (const float4*)src + (size_t)idx * 2;
  float4 a = s[0], b = s[1];
  union { unsigned short u[8]; uint4 v; } o;
  o.u[0] = f2bf(a.x); o.u[1] = f2bf(a.y); o.u[2] = f2bf(a.z); o.u[3] = f2bf(a.w);
  o.u[4] = f2bf(b.x); o.u[5] = f2bf(b.y); o.u[6] = f2bf(b.z); o.u[7] = f2bf(b.w);
  ((uint4*)dst)[idx] = o.v;
}

// ------- prep kernel 2: values_o (bh,n,d) f32 -> bf16 transposed to (bh,d,n) -------
__global__ __launch_bounds__(256)
void transpose_cvt_kernel(const float* __restrict__ vals, unsigned short* __restrict__ vt) {
  const int row  = blockIdx.x;          // bh*DIM + dcol  (2048 rows)
  const int bh   = row >> 6;
  const int dcol = row & 63;
  const int j0   = threadIdx.x * 8;     // 256 threads x 8 = 2048 = SEQ
  union { unsigned short u[8]; uint4 v; } o;
#pragma unroll
  for (int e = 0; e < 8; ++e)
    o.u[e] = f2bf(vals[((size_t)bh * SEQ + j0 + e) * DIM + dcol]);
  *(uint4*)(vt + (size_t)row * SEQ + j0) = o.v;
}

// --------------------------------- main kernel ---------------------------------
__global__ __launch_bounds__(256)
void pa_flash_kernel(const float* __restrict__ q, const float* __restrict__ k,
                     const float* __restrict__ v,
                     const float* __restrict__ ko,             // keys_o f32 (mask dot)
                     const float* __restrict__ vo,             // values_o f32 (mask fix)
                     const unsigned short* __restrict__ kbf,   // keys_o bf16 (bh,n,d)
                     const unsigned short* __restrict__ vtb,   // values_o bf16 (bh,d,n)
                     float* __restrict__ out)
{
  // Per-wave P-transpose staging: 16 rows x 32 bf16, row stride 40 (80B, 16B aligned).
  __shared__ __align__(16) unsigned short Plds[8][16][40];

  const int lane = threadIdx.x & 31;
  const int wave = threadIdx.x >> 5;
  const int nlo  = lane & 15;   // column/row-in-tile index
  const int hh   = lane >> 4;   // half-wave select

  const int bh   = blockIdx.x >> 4;       // 0..31  (b*h)
  const int grp  = blockIdx.x & 15;       // 0..15  (query tile group)
  const int bb   = bh >> 4;
  const int head = bh & 15;
  const int i0   = (grp * 8 + wave) * 16; // first query row of this wave

  const size_t hoff = (size_t)bh * SEQ * DIM;
  const float* Qh  = q  + hoff;
  const float* Kh  = k  + hoff;
  const float* Vh  = v  + hoff;
  const float* KOh = ko + hoff;
  const float* VOh = vo + hoff;
  const unsigned short* Kb = kbf + hoff;
  const unsigned short* Vt = vtb + hoff;  // (bh,d,n): same element count per head

  const float kscale = 0.125f * 1.44269504088896340736f; // d^-0.5 * log2(e)

  // ---- Q as two bf16 A-fragments, scale folded in (done once per wave) ----
  Frag16 qf[2];
  {
    const float* qrow = Qh + (size_t)(i0 + nlo) * DIM;
#pragma unroll
    for (int c = 0; c < 2; ++c) {
      pack8s(qrow + c*32 + hh*8,      &qf[c].u[0], kscale);
      pack8s(qrow + c*32 + 16 + hh*8, &qf[c].u[8], kscale);
    }
  }

  // ---- all-ones bf16 B fragment: row-sum of P via WMMA (softmax denominator) ----
  Frag16 onesf;
#pragma unroll
  for (int e = 0; e < 16; ++e) onesf.u[e] = 0x3F80u;  // bf16 1.0

  // ---- prologue dots (log2 domain):
  //   sd_i    = (q_i . k_i)          * scale  -> superdiag logit (weights v_i)
  //   smask_i = (q_i . keys_o[i+1])  * scale  -> logit of the REPLACED column i+1,
  //                                              subtracted as a rank-1 fix at the end
  float sd_all, sm_all;
  {
    const int   row  = i0 + nlo;
    const int   nrow = (row + 1 < SEQ) ? row + 1 : SEQ - 1;  // clamped (unused if i==n-1)
    const float4* qr = (const float4*)(Qh  + (size_t)row  * DIM + hh*32);
    const float4* kr = (const float4*)(Kh  + (size_t)row  * DIM + hh*32);
    const float4* mr = (const float4*)(KOh + (size_t)nrow * DIM + hh*32);
    float p0 = 0.f, p1 = 0.f;
#pragma unroll
    for (int t = 0; t < 8; ++t) {
      float4 a = qr[t], b = kr[t], c = mr[t];
      p0 += a.x*b.x + a.y*b.y + a.z*b.z + a.w*b.w;
      p1 += a.x*c.x + a.y*c.y + a.z*c.z + a.w*c.w;
    }
    p0 += __shfl_xor(p0, 16);
    p1 += __shfl_xor(p1, 16);
    sd_all = p0 * kscale;   // lanes x and x+16 hold values for row i0+x
    sm_all = p1 * kscale;
  }

  float sd2[8], sm2[8], m_run[8];
#pragma unroll
  for (int r = 0; r < 8; ++r) {
    sd2[r]   = __shfl(sd_all, r + 8*hh);  // row m = r + 8*hh of this tile
    sm2[r]   = __shfl(sm_all, r + 8*hh);
    m_run[r] = sd2[r];                    // superdiag seeds the running max
  }

  v8f acc[4], lsum;
#pragma unroll
  for (int dt = 0; dt < 4; ++dt) acc[dt] = (v8f){0.f,0.f,0.f,0.f,0.f,0.f,0.f,0.f};
  lsum = (v8f){0.f,0.f,0.f,0.f,0.f,0.f,0.f,0.f};

  for (int j0 = 0; j0 < SEQ; j0 += 32) {
    // Branch-free prefetch of the next key/value block (clamped; non-faulting).
    const int jp = (j0 + 32 < SEQ) ? j0 + 32 : j0;
    __builtin_prefetch(Kb + (size_t)(jp + nlo) * DIM, 0, 1);
    __builtin_prefetch(Vt + (size_t)nlo * SEQ + jp, 0, 1);

    // ---- K block as B fragments (pure b128 loads) ----
    Frag16 kf[2][2];
#pragma unroll
    for (int t = 0; t < 2; ++t) {
      const unsigned short* krow = Kb + (size_t)(j0 + t*16 + nlo) * DIM;
#pragma unroll
      for (int c = 0; c < 2; ++c) {
        kf[t][c].q[0] = *(const uint4*)(krow + c*32 + hh*8);
        kf[t][c].q[1] = *(const uint4*)(krow + c*32 + 16 + hh*8);
      }
    }

    // ---- S = (Q K^T) for two 16x16 key tiles (logits pre-scaled via Q) ----
    v8f s[2];
#pragma unroll
    for (int t = 0; t < 2; ++t) {
      v8f c0 = (v8f){0.f,0.f,0.f,0.f,0.f,0.f,0.f,0.f};
      c0 = __builtin_amdgcn_wmma_f32_16x16x32_bf16(false, qf[0].v, false, kf[t][0].v,
                                                   (short)0, c0, false, false);
      c0 = __builtin_amdgcn_wmma_f32_16x16x32_bf16(false, qf[1].v, false, kf[t][1].v,
                                                   (short)0, c0, false, false);
      s[t] = c0;
    }

    // ---- per-row max, packed two-rows-per-register in bf16 (m only needs to
    //      approximate the max; bf16 rounding is harmless).  4 butterfly stages
    //      on 4 packed registers: 16 bpermutes instead of 32, pk_max VALU. ----
    float mxp[8];
#pragma unroll
    for (int r = 0; r < 8; ++r) mxp[r] = fmaxf(s[0][r], s[1][r]);
    unsigned pk[4];
#pragma unroll
    for (int p = 0; p < 4; ++p) pk[p] = pk_bf16(mxp[2*p], mxp[2*p + 1]);
#pragma unroll
    for (int stage = 1; stage <= 8; stage <<= 1) {
      unsigned sh[4];
#pragma unroll
      for (int p = 0; p < 4; ++p) sh[p] = __shfl_xor(pk[p], stage);
#pragma unroll
      for (int p = 0; p < 4; ++p) pk[p] = pk_max_bf16(pk[p], sh[p]);
    }
    float mx[8];
#pragma unroll
    for (int p = 0; p < 4; ++p) {
      mx[2*p]     = __builtin_bit_cast(float, pk[p] << 16);
      mx[2*p + 1] = __builtin_bit_cast(float, pk[p] & 0xFFFF0000u);
    }

    // ---- rescale only when the running max actually moves (wave-uniform:
    //      mx is uniform across each half-wave, ballot covers all 16 rows) ----
    float upd = 0.f;
#pragma unroll
    for (int r = 0; r < 8; ++r) upd = fmaxf(upd, mx[r] - m_run[r]);
    if (__any(upd > 0.f)) {
      float alpha[8];
#pragma unroll
      for (int r = 0; r < 8; ++r) {
        const float mnew = fmaxf(m_run[r], mx[r]);
        alpha[r] = __builtin_amdgcn_exp2f(m_run[r] - mnew);
        m_run[r] = mnew;
      }
#pragma unroll
      for (int r = 0; r < 8; ++r) {
        lsum[r] *= alpha[r];
#pragma unroll
        for (int dt = 0; dt < 4; ++dt) acc[dt][r] *= alpha[r];
      }
    }

    // ---- probabilities (unmasked; replaced column fixed up in the epilogue) ----
#pragma unroll
    for (int r = 0; r < 8; ++r)
#pragma unroll
      for (int t = 0; t < 2; ++t)
        s[t][r] = __builtin_amdgcn_exp2f(s[t][r] - m_run[r]);

    // ---- P: C layout -> row-major bf16 in per-wave LDS (DS ops in-order per wave) ----
#pragma unroll
    for (int t = 0; t < 2; ++t)
#pragma unroll
      for (int r = 0; r < 8; ++r)
        Plds[wave][r + 8*hh][t*16 + nlo] = f2bf(s[t][r]);

    // ---- reload P as a 16x32 A-fragment ----
    Frag16 pf;
    {
      const unsigned short* prow = &Plds[wave][nlo][0];
      pf.q[0] = *(const uint4*)(prow + hh*8);        // j = 8*hh + 0..7
      pf.q[1] = *(const uint4*)(prow + 16 + hh*8);   // j = 16 + 8*hh + 0..7
    }

    // ---- O += P @ V (4 d-tiles) and row-sum(P) via the ones fragment ----
#pragma unroll
    for (int dt = 0; dt < 4; ++dt) {
      Frag16 vf;  // B fragment: elem e -> key k = 8*hh + e (e<8), 16 + 8*hh + (e-8) (e>=8), col = dt*16 + nlo
      const unsigned short* vrow = Vt + (size_t)(dt*16 + nlo) * SEQ + j0 + hh*8;
      vf.q[0] = *(const uint4*)(vrow);
      vf.q[1] = *(const uint4*)(vrow + 16);
      acc[dt] = __builtin_amdgcn_wmma_f32_16x16x32_bf16(false, pf.v, false, vf.v,
                                                        (short)0, acc[dt], false, false);
    }
    lsum = __builtin_amdgcn_wmma_f32_16x16x32_bf16(false, pf.v, false, onesf.v,
                                                   (short)0, lsum, false, false);
  }

  // ---- epilogue: superdiag add, pad zero-logit, rank-1 removal of column i+1 ----
#pragma unroll
  for (int r = 0; r < 8; ++r) {
    const int i   = i0 + r + 8*hh;
    const float w = __builtin_amdgcn_exp2f(sd2[r] - m_run[r]);     // superdiag weight
    float l  = lsum[r] + w;                                        // all keys + superdiag
    float wm = 0.f;
    if (i < SEQ - 1) {
      wm = __builtin_amdgcn_exp2f(sm2[r] - m_run[r]);              // replaced column
      l += __builtin_amdgcn_exp2f(-m_run[r]);                      // pad zero-logit
      l -= wm;
    }
    const float inv = 1.0f / l;
    const int   ni  = (i + 1 < SEQ) ? i + 1 : i;                   // clamped (wm==0 then)
    const float* vorow = VOh + (size_t)ni * DIM;
#pragma unroll
    for (int dt = 0; dt < 4; ++dt) {
      const int dc = dt*16 + nlo;
      const float vv = Vh[(size_t)i * DIM + dc];
      out[(((size_t)bb * SEQ + i) * HEADS + head) * DIM + dc] =
          (acc[dt][r] - wm * vorow[dc] + w * vv) * inv;
    }
  }
}

extern "C" void kernel_launch(void* const* d_in, const int* in_sizes, int n_in,
                              void* d_out, int out_size, void* d_ws, size_t ws_size,
                              hipStream_t stream) {
  const float* q  = (const float*)d_in[0];
  const float* k  = (const float*)d_in[1];
  const float* v  = (const float*)d_in[2];
  const float* ko = (const float*)d_in[3];
  const float* vo = (const float*)d_in[4];
  float* out = (float*)d_out;

  const size_t nElem = (size_t)BATCH * HEADS * SEQ * DIM;   // 4,194,304
  unsigned short* kbf = (unsigned short*)d_ws;              // 8 MiB
  unsigned short* vtb = kbf + nElem;                        // 8 MiB  (16 MiB total ws)

  // Prep: bf16 copies of keys_o (same layout) and values_o (transposed per head).
  cvt_bf16_kernel<<<dim3((unsigned)(nElem / (256 * 8))), 256, 0, stream>>>(ko, kbf);
  transpose_cvt_kernel<<<dim3(BATCH * HEADS * DIM), 256, 0, stream>>>(vo, vtb);

  // 32 (b*h) heads x 16 groups of 128 query rows; 256 threads = 8 wave32 per block.
  pa_flash_kernel<<<dim3(BATCH * HEADS * (SEQ / 128)), 256, 0, stream>>>(
      q, k, v, ko, vo, kbf, vtb, out);
}
